// Speller_14044543058558
// MI455X (gfx1250) — matrix operand order repaired and checked by reading, not compiled, not run on hardware
//
#include <hip/hip_runtime.h>
#include <math.h>

// ---------------------------------------------------------------------------
// Speller (LAS decoder) for MI455X / gfx1250, wave32 + WMMA bf16.
//   - bf16 v_wmma_f32_16x16x32_bf16 for all recurrent GEMMs and the one-time
//     key/value projection (f32 accumulate).
//   - attention / softmax / logits kept in f32 VALU (batched matvec, L2-resident).
//   - embedding contribution to LSTM0 gates precomputed as a 30-row table.
// ---------------------------------------------------------------------------

typedef unsigned short bf16_t;                                     // storage
typedef __attribute__((ext_vector_type(16))) __bf16 bf16x16;
typedef __attribute__((ext_vector_type(8)))  __bf16 bf16x8;
typedef __attribute__((ext_vector_type(8)))  float  f32x8;

#define B_      128
#define T_ENC_  400
#define ENC_D_  512
#define T_DEC_  300
#define VOCAB_  30
#define HID0_   512
#define HID1_   256
#define PROJ_   256
#define KCAT_   768   // [ctx(256)|h0(512)]  and  [h0(512)|h1(256)]

__device__ __forceinline__ bf16_t f2bf(float f) {       // f32 -> bf16 (RNE)
  union { float f; unsigned u; } v; v.f = f;
  unsigned r = v.u + 0x7fffu + ((v.u >> 16) & 1u);
  return (bf16_t)(r >> 16);
}

// A fragment, 16x32 bf16: lane<16 -> K {0..15}, lane>=16 -> K {16..31} (matches B)
__device__ __forceinline__ bf16x16 load_a_bf16(const bf16_t* X, int ld, int m0, int k0, int lane) {
  const int row = m0 + (lane & 15);
  const int kb  = k0 + ((lane >> 4) << 4);
  const bf16_t* p = X + row * ld + kb;
  bf16x8 lo = *(const bf16x8*)(p);
  bf16x8 hi = *(const bf16x8*)(p + 8);
  bf16x16 r;
#pragma unroll
  for (int i = 0; i < 8; ++i) { r[i] = lo[i]; r[i + 8] = hi[i]; }
  return r;
}

// Same A fragment but converting from an f32 source on the fly.
__device__ __forceinline__ bf16x16 load_a_f32(const float* X, int ld, int m0, int k0, int lane) {
  const int row = m0 + (lane & 15);
  const int kb  = k0 + ((lane >> 4) << 4);
  const float* p = X + row * ld + kb;
  f32x8 lo = *(const f32x8*)(p);
  f32x8 hi = *(const f32x8*)(p + 8);
  bf16x16 r;
#pragma unroll
  for (int i = 0; i < 8; ++i) { r[i] = (__bf16)lo[i]; r[i + 8] = (__bf16)hi[i]; }
  return r;
}

// B fragment, 32x16 bf16, weights stored row-major [N][K]: lane = column,
// contiguous 16-element K run (lanes 16..31 offset +16).
__device__ __forceinline__ bf16x16 load_b_bf16(const bf16_t* W, int ld, int n0, int k0, int lane) {
  const int col = n0 + (lane & 15);
  const int kb  = k0 + ((lane >> 4) << 4);
  return *(const bf16x16*)(W + col * ld + kb);
}

__device__ __forceinline__ f32x8 wmma_bf16(bf16x16 a, bf16x16 b, f32x8 c) {
  return __builtin_amdgcn_wmma_f32_16x16x32_bf16(false, a, false, b, (short)0, c, false, false);
}

__device__ __forceinline__ float sigm(float x) { return 1.f / (1.f + __expf(-x)); }

// ---------------------------------------------------------------------------
// One-time prep: pack/transpose weights to bf16, zero recurrent state.
// ---------------------------------------------------------------------------
__global__ void prep_pack(const float* __restrict__ Wi0, const float* __restrict__ Wh0,
                          const float* __restrict__ Wi1, const float* __restrict__ Wh1,
                          const float* __restrict__ Wk,  const float* __restrict__ Wv,
                          const float* __restrict__ bi1, const float* __restrict__ bh1,
                          bf16_t* __restrict__ W0, bf16_t* __restrict__ W1,
                          bf16_t* __restrict__ WkT, bf16_t* __restrict__ WvT,
                          float* __restrict__ b1sum, float* __restrict__ c0,
                          float* __restrict__ c1, bf16_t* __restrict__ a0_0,
                          bf16_t* __restrict__ a1_0)
{
  const long N0 = 2048l * 768, N1 = 1024l * 768, NT = 256l * 512;
  const long C0 = N0, C1 = C0 + N1, C2 = C1 + NT, C3 = C2 + NT;
  const long C4 = C3 + 1024;
  const long C5 = C4 + 128l * 512;
  const long C6 = C5 + 128l * 256;
  const long C7 = C6 + 128l * 768;
  const long C8 = C7 + 128l * 768;
  for (long i = (long)blockIdx.x * 256 + threadIdx.x; i < C8; i += (long)gridDim.x * 256) {
    if (i < C0) {            // W0 [2048][768] = [Wi0_ctx(256) | Wh0(512)]
      long n = i / 768, k = i % 768;
      float v = (k < 256) ? Wi0[n * 768 + 512 + k] : Wh0[n * 512 + (k - 256)];
      W0[i] = f2bf(v);
    } else if (i < C1) {     // W1 [1024][768] = [Wi1(512) | Wh1(256)]
      long j = i - C0, n = j / 768, k = j % 768;
      float v = (k < 512) ? Wi1[n * 512 + k] : Wh1[n * 256 + (k - 512)];
      W1[j] = f2bf(v);
    } else if (i < C2) {     // WkT [256][512]
      long j = i - C1, n = j / 512, k = j % 512;
      WkT[j] = f2bf(Wk[k * 256 + n]);
    } else if (i < C3) {     // WvT [256][512]
      long j = i - C2, n = j / 512, k = j % 512;
      WvT[j] = f2bf(Wv[k * 256 + n]);
    } else if (i < C4) {     // bi1 + bh1
      long j = i - C3; b1sum[j] = bi1[j] + bh1[j];
    } else if (i < C5) { c0[i - C4] = 0.f; }
    else if (i < C6) { c1[i - C5] = 0.f; }
    else if (i < C7) { a0_0[i - C6] = 0; }
    else             { a1_0[i - C7] = 0; }
  }
}

// Gemb[v][n] = emb[v] . Wi0_emb[n] + bi0[n] + bh0[n]   ([30][2048], one-time)
__global__ void prep_gemb(const float* __restrict__ emb, const float* __restrict__ Wi0,
                          const float* __restrict__ bi0, const float* __restrict__ bh0,
                          float* __restrict__ Gemb)
{
  const int g = blockIdx.x * 256 + threadIdx.x;
  if (g >= VOCAB_ * 2048) return;
  const int v = g >> 11, n = g & 2047;
  float acc = bi0[n] + bh0[n];
  const float* e = emb + v * 512;
  const float* w = Wi0 + n * 768;            // first 512 cols = embed part
  for (int k = 0; k < 512; ++k) acc += e[k] * w[k];
  Gemb[g] = acc;
}

// keys/values projection: out[51200][256] = enc[51200][512] @ WT.T + bias (WMMA)
__global__ void kv_gemm(const float* __restrict__ A, const bf16_t* __restrict__ WT,
                        const float* __restrict__ bias, float* __restrict__ out)
{
  const int lane = threadIdx.x;
  const int n0 = blockIdx.x << 4;
  const int m0 = blockIdx.y << 4;
  f32x8 acc = {};
#pragma unroll 4
  for (int k = 0; k < ENC_D_; k += 32) {
    bf16x16 a = load_a_f32(A, ENC_D_, m0, k, lane);
    bf16x16 b = load_b_bf16(WT, ENC_D_, n0, k, lane);
    acc = wmma_bf16(a, b, acc);
  }
  const int nc = n0 + (lane & 15);
  const int mb = m0 + ((lane >> 4) << 3);
  const float bb = bias[nc];
#pragma unroll
  for (int r = 0; r < 8; ++r) out[(long)(mb + r) * PROJ_ + nc] = acc[r] + bb;
}

// ---------------------------------------------------------------------------
// Per-step kernels
// ---------------------------------------------------------------------------
// LSTM0: gates = [ctx|h0] @ W0.T + Gemb[y_in] ; fused cell nonlinearity.
__global__ void lstm0_step(const bf16_t* __restrict__ a0, const bf16_t* __restrict__ W0,
                           const float* __restrict__ Gemb, const int* __restrict__ y,
                           float* __restrict__ c0, bf16_t* __restrict__ a0n,
                           bf16_t* __restrict__ a1p, int t)
{
  const int lane = threadIdx.x;
  const int n0 = blockIdx.x << 4;          // hidden block in [0,512)
  const int m0 = blockIdx.y << 4;          // batch block
  f32x8 ai = {}, af = {}, ag = {}, ao = {};
  for (int k = 0; k < KCAT_; k += 32) {
    bf16x16 a  = load_a_bf16(a0, KCAT_, m0, k, lane);
    bf16x16 bi = load_b_bf16(W0, KCAT_, n0,        k, lane);
    bf16x16 bf = load_b_bf16(W0, KCAT_, 512 + n0,  k, lane);
    bf16x16 bg = load_b_bf16(W0, KCAT_, 1024 + n0, k, lane);
    bf16x16 bo = load_b_bf16(W0, KCAT_, 1536 + n0, k, lane);
    ai = wmma_bf16(a, bi, ai);
    af = wmma_bf16(a, bf, af);
    ag = wmma_bf16(a, bg, ag);
    ao = wmma_bf16(a, bo, ao);
  }
  const int hn  = n0 + (lane & 15);
  const int mb0 = m0 + ((lane >> 4) << 3);
#pragma unroll
  for (int r = 0; r < 8; ++r) {
    const int mb = mb0 + r;
    const int yv = (t == 0) ? 0 : y[mb * T_DEC_ + (t - 1)];   // SOS at t=0
    const float* gE = Gemb + yv * 2048;
    const float gi = ai[r] + gE[hn];
    const float gf = af[r] + gE[512 + hn];
    const float gg = ag[r] + gE[1024 + hn];
    const float go = ao[r] + gE[1536 + hn];
    const float c  = sigm(gf) * c0[mb * HID0_ + hn] + sigm(gi) * tanhf(gg);
    const float h  = sigm(go) * tanhf(c);
    c0[mb * HID0_ + hn] = c;
    const bf16_t hb = f2bf(h);
    a0n[mb * KCAT_ + PROJ_ + hn] = hb;     // next-step [ctx | h0]
    a1p[mb * KCAT_ + hn]         = hb;     // this-step [h0 | h1]
  }
}

// LSTM1: gates = [h0|h1] @ W1.T + (bi1+bh1) ; fused cell nonlinearity.
__global__ void lstm1_step(const bf16_t* __restrict__ a1, const bf16_t* __restrict__ W1,
                           const float* __restrict__ b1sum, float* __restrict__ c1,
                           bf16_t* __restrict__ a1n, float* __restrict__ h1f)
{
  const int lane = threadIdx.x;
  const int n0 = blockIdx.x << 4;          // hidden block in [0,256)
  const int m0 = blockIdx.y << 4;
  f32x8 ai = {}, af = {}, ag = {}, ao = {};
  for (int k = 0; k < KCAT_; k += 32) {
    bf16x16 a  = load_a_bf16(a1, KCAT_, m0, k, lane);
    bf16x16 bi = load_b_bf16(W1, KCAT_, n0,       k, lane);
    bf16x16 bf = load_b_bf16(W1, KCAT_, 256 + n0, k, lane);
    bf16x16 bg = load_b_bf16(W1, KCAT_, 512 + n0, k, lane);
    bf16x16 bo = load_b_bf16(W1, KCAT_, 768 + n0, k, lane);
    ai = wmma_bf16(a, bi, ai);
    af = wmma_bf16(a, bf, af);
    ag = wmma_bf16(a, bg, ag);
    ao = wmma_bf16(a, bo, ao);
  }
  const int hn  = n0 + (lane & 15);
  const int mb0 = m0 + ((lane >> 4) << 3);
#pragma unroll
  for (int r = 0; r < 8; ++r) {
    const int mb = mb0 + r;
    const float gi = ai[r] + b1sum[hn];
    const float gf = af[r] + b1sum[256 + hn];
    const float gg = ag[r] + b1sum[512 + hn];
    const float go = ao[r] + b1sum[768 + hn];
    const float c  = sigm(gf) * c1[mb * HID1_ + hn] + sigm(gi) * tanhf(gg);
    const float h  = sigm(go) * tanhf(c);
    c1[mb * HID1_ + hn] = c;
    h1f[mb * HID1_ + hn] = h;                    // f32 for q/logits
    a1n[mb * KCAT_ + HID0_ + hn] = f2bf(h);      // next-step [h0 | h1]
  }
}

// Fused per-batch-element: q = h1@Wq+bq, masked softmax attention, ctx, logits.
__global__ void attn_step(const float* __restrict__ keys, const float* __restrict__ values,
                          const float* __restrict__ Wq, const float* __restrict__ bq,
                          const int* __restrict__ lens, const float* __restrict__ h1f,
                          const float* __restrict__ emb, const float* __restrict__ b_out,
                          bf16_t* __restrict__ a0n, float* __restrict__ out, int t)
{
  __shared__ float h1s[256];
  __shared__ float qs[256];
  __shared__ float sK[32 * 257];      // keys tile, padded stride -> conflict-free
  __shared__ float E[400];
  __shared__ float part[8][32];
  __shared__ float red[256];
  __shared__ float ctxs[256];
  const int b = blockIdx.x;
  const int tid = threadIdx.x;
  const int len = lens[b];

  h1s[tid] = h1f[b * 256 + tid];
  __syncthreads();

  // q = h1 @ Wq + bq   (Wq is [256][256] k-major: coalesced across lanes)
  float qa = bq[tid];
  for (int k = 0; k < 256; ++k) qa += h1s[k] * Wq[k * 256 + tid];
  qs[tid] = qa;
  __syncthreads();

  // energy = (keys . q) * 1/sqrt(256), masked
  const float scale = 0.0625f;
  for (int tc = 0; tc < T_ENC_; tc += 32) {
    const int cs = (T_ENC_ - tc < 32) ? (T_ENC_ - tc) : 32;
    for (int idx = tid; idx < cs * 256; idx += 256)
      sK[(idx >> 8) * 257 + (idx & 255)] = keys[((long)(b * T_ENC_ + tc) << 8) + idx];
    __syncthreads();
    const int tl = tid & 31, kg = tid >> 5;     // 8 k-slices x 32 timesteps
    float p = 0.f;
    const float* kr = &sK[tl * 257 + kg * 32];
    const float* qr = &qs[kg * 32];
    for (int k = 0; k < 32; ++k) p += kr[k] * qr[k];
    part[kg][tl] = p;
    __syncthreads();
    if (tid < cs) {
      float e = 0.f;
      for (int g = 0; g < 8; ++g) e += part[g][tid];
      E[tc + tid] = (tc + tid < len) ? e * scale : -1e9f;
    }
    __syncthreads();
  }

  // softmax over 400
  float lm = -1e30f;
  for (int i = tid; i < T_ENC_; i += 256) lm = fmaxf(lm, E[i]);
  red[tid] = lm; __syncthreads();
  for (int s = 128; s > 0; s >>= 1) { if (tid < s) red[tid] = fmaxf(red[tid], red[tid + s]); __syncthreads(); }
  const float mx = red[0]; __syncthreads();
  float ls = 0.f;
  for (int i = tid; i < T_ENC_; i += 256) { float w = __expf(E[i] - mx); E[i] = w; ls += w; }
  red[tid] = ls; __syncthreads();
  for (int s = 128; s > 0; s >>= 1) { if (tid < s) red[tid] += red[tid + s]; __syncthreads(); }
  const float inv = 1.f / red[0];
  __syncthreads();

  // ctx[p] = sum_t attn[t] * values[b,t,p]   (coalesced across lanes)
  float cacc = 0.f;
  for (int tt = 0; tt < T_ENC_; ++tt)
    cacc += E[tt] * values[((long)(b * T_ENC_ + tt) << 8) + tid];
  const float ctx = cacc * inv;
  ctxs[tid] = ctx;
  a0n[b * KCAT_ + tid] = f2bf(ctx);            // next-step [ctx | h0]
  __syncthreads();

  // logits = [h1 | ctx] @ emb.T + b_out  (weight-tied; only 30 outputs)
  if (tid < VOCAB_) {
    float acc = b_out[tid];
    const float* ev = emb + tid * 512;
    for (int k = 0; k < 256; ++k) acc += h1s[k] * ev[k];
    for (int k = 0; k < 256; ++k) acc += ctxs[k] * ev[256 + k];
    out[((long)b * T_DEC_ + t) * VOCAB_ + tid] = acc;
  }
}

// ---------------------------------------------------------------------------
extern "C" void kernel_launch(void* const* d_in, const int* in_sizes, int n_in,
                              void* d_out, int out_size, void* d_ws, size_t ws_size,
                              hipStream_t stream) {
  (void)in_sizes; (void)n_in; (void)out_size; (void)ws_size;
  const float* enc   = (const float*)d_in[0];
  const int*   lens  = (const int*)  d_in[1];
  const int*   y     = (const int*)  d_in[2];
  const float* emb   = (const float*)d_in[3];
  const float* Wk    = (const float*)d_in[4];
  const float* bk    = (const float*)d_in[5];
  const float* Wv    = (const float*)d_in[6];
  const float* bv    = (const float*)d_in[7];
  const float* Wq    = (const float*)d_in[8];
  const float* bq    = (const float*)d_in[9];
  const float* Wi0   = (const float*)d_in[10];
  const float* Wh0   = (const float*)d_in[11];
  const float* bi0   = (const float*)d_in[12];
  const float* bh0   = (const float*)d_in[13];
  const float* Wi1   = (const float*)d_in[14];
  const float* Wh1   = (const float*)d_in[15];
  const float* bi1   = (const float*)d_in[16];
  const float* bh1   = (const float*)d_in[17];
  const float* b_out = (const float*)d_in[18];
  float* out = (float*)d_out;

  // workspace carve (256B aligned) -- total ~112 MB
  char* wp = (char*)d_ws;
  auto alloc = [&](size_t bytes) -> void* {
    void* r = (void*)wp; wp += (bytes + 255) & ~(size_t)255; return r;
  };
  bf16_t* W0    = (bf16_t*)alloc(2048ull * 768 * 2);
  bf16_t* W1    = (bf16_t*)alloc(1024ull * 768 * 2);
  bf16_t* WkT   = (bf16_t*)alloc(256ull * 512 * 2);
  bf16_t* WvT   = (bf16_t*)alloc(256ull * 512 * 2);
  float*  Gemb  = (float*) alloc(30ull * 2048 * 4);
  float*  b1sum = (float*) alloc(1024ull * 4);
  float*  keys  = (float*) alloc((size_t)B_ * T_ENC_ * PROJ_ * 4);
  float*  vals  = (float*) alloc((size_t)B_ * T_ENC_ * PROJ_ * 4);
  float*  c0    = (float*) alloc((size_t)B_ * HID0_ * 4);
  float*  c1    = (float*) alloc((size_t)B_ * HID1_ * 4);
  float*  h1f   = (float*) alloc((size_t)B_ * HID1_ * 4);
  bf16_t* a0    = (bf16_t*)alloc(2ull * B_ * KCAT_ * 2);   // ping-pong [ctx|h0]
  bf16_t* a1    = (bf16_t*)alloc(2ull * B_ * KCAT_ * 2);   // ping-pong [h0|h1]
  const size_t SA = (size_t)B_ * KCAT_;

  // one-time prep
  {
    const long total = 2048l*768 + 1024l*768 + 2*(256l*512) + 1024
                     + 128l*512 + 128l*256 + 2*(128l*768);
    const int blocks = (int)((total + 255) / 256);
    prep_pack<<<blocks, 256, 0, stream>>>(Wi0, Wh0, Wi1, Wh1, Wk, Wv, bi1, bh1,
                                          W0, W1, WkT, WvT, b1sum, c0, c1,
                                          a0, a1);
  }
  prep_gemb<<<(VOCAB_ * 2048 + 255) / 256, 256, 0, stream>>>(emb, Wi0, bi0, bh0, Gemb);
  kv_gemm<<<dim3(PROJ_ / 16, (B_ * T_ENC_) / 16), 32, 0, stream>>>(enc, WkT, bk, keys);
  kv_gemm<<<dim3(PROJ_ / 16, (B_ * T_ENC_) / 16), 32, 0, stream>>>(enc, WvT, bv, vals);

  // 300 sequential decode steps
  for (int t = 0; t < T_DEC_; ++t) {
    const int p = t & 1;
    lstm0_step<<<dim3(HID0_ / 16, B_ / 16), 32, 0, stream>>>(
        a0 + p * SA, W0, Gemb, y, c0, a0 + (1 - p) * SA, a1 + p * SA, t);
    lstm1_step<<<dim3(HID1_ / 16, B_ / 16), 32, 0, stream>>>(
        a1 + p * SA, W1, b1sum, c1, a1 + (1 - p) * SA, h1f);
    attn_step<<<B_, 256, 0, stream>>>(
        keys, vals, Wq, bq, lens, h1f, emb, b_out, a0 + (1 - p) * SA, out, t);
  }
}
